// Decoder_57561151701599
// MI455X (gfx1250) — matrix operand order repaired and checked by reading, not compiled, not run on hardware
//
#include <hip/hip_runtime.h>
#include <hip/hip_fp16.h>
#include <hip/hip_bf16.h>

// ---------------------------------------------------------------------------
// MI455X (gfx1250) decoder block: self-attn + cross-attn + MLP.
// Compute-bound (~275 GFLOP vs ~5us of HBM traffic at 23.3 TB/s): every
// matmul runs on v_wmma_f32_16x16x32_f16 (wave32 WMMA). Tile staging uses the
// Tensor Data Mover (tensor_load_to_lds, TENSORcnt) with LDS double-buffering
// so DMA for tile t+1 overlaps the WMMA chain for tile t.
// ---------------------------------------------------------------------------

typedef __attribute__((ext_vector_type(16))) _Float16 v16h;
typedef __attribute__((ext_vector_type(8)))  _Float16 v8h;
typedef __attribute__((ext_vector_type(8)))  float    v8f;
typedef __attribute__((ext_vector_type(4)))  unsigned v4u;
typedef __attribute__((ext_vector_type(8)))  int      v8i;
typedef __attribute__((ext_vector_type(4)))  int      v4i;

#define DEVI __device__ __forceinline__

#if __has_builtin(__builtin_amdgcn_tensor_load_to_lds) && \
    __has_builtin(__builtin_amdgcn_s_wait_tensorcnt)
#define USE_TDM 1
#else
#define USE_TDM 0
#endif

static constexpr int B_   = 2;
static constexpr int N_   = 2048;
static constexpr int C_   = 1024;
static constexpr int H_   = 16;
static constexpr int HD_  = 64;
static constexpr int Y_   = 1024;
static constexpr int HID_ = 4096;
static constexpr int M_   = B_ * N_;          // 4096 rows in the token dim
static constexpr float SM_SCALE = 0.125f;     // HD^-0.5

// ---------------- WMMA helpers (wave32, 16x16x32 f16 -> f32) ----------------

DEVI v8f wmma_f16(v16h a, v16h b, v8f c) {
  // 8 args: (neg_a, A, neg_b, B, c_mod, C, reuse_a, reuse_b)
  return __builtin_amdgcn_wmma_f32_16x16x32_f16(false, a, false, b, (short)0, c,
                                                false, false);
}

// A-matrix 16x32 f16 fragment from LDS (row-major, ld in elements).
// Lanes 0-15: row = lane, K = 0..7 and 16..23 ; lanes 16-31: row = lane-16,
// K = 8..15 and 24..31  (per CDNA5 ISA 7.12.2).
DEVI v16h ld_afrag(const _Float16* base, int ld, int lane) {
  const int r  = lane & 15;
  const int kh = lane >> 4;
  const _Float16* p = base + r * ld + kh * 8;
  v8h lo = *(const v8h*)p;          // K = kh*8 .. +7
  v8h hi = *(const v8h*)(p + 16);   // K = 16+kh*8 .. +7
  v16h o;
#pragma unroll
  for (int i = 0; i < 8; ++i) { o[i] = lo[i]; o[8 + i] = hi[i]; }
  return o;
}

// B-matrix 32x16 f16 fragment from a row-major [n][k] LDS tile: lane&15 = n,
// lanes 0-15: K = 0..7 and 16..23; lanes 16-31: K = 8..15 and 24..31.
DEVI v16h ld_bfrag(const _Float16* base, int ld, int lane) {
  const int n   = lane & 15;
  const int off = (lane >> 4) * 8;
  const _Float16* p = base + n * ld + off;
  v8h lo = *(const v8h*)p;
  v8h hi = *(const v8h*)(p + 16);
  v16h o;
#pragma unroll
  for (int i = 0; i < 8; ++i) { o[i] = lo[i]; o[8 + i] = hi[i]; }
  return o;
}

DEVI float gelu_exact(float x) {
  return 0.5f * x * (1.0f + erff(x * 0.70710678118654752f));
}

// -------------------- Tensor Data Mover 2D tile descriptor ------------------
// Loads a [tile_rows x tile_k] f16 tile (row stride = stride_elems) from
// global into LDS at lds_addr, inserting pad_amt dwords of LDS padding every
// pad_int-coded interval so the LDS rows keep the bank-conflict-free stride.
// D# layout per CDNA5 ISA 8.3/8.4 (group0: flags/lds/global/type=2,
// group1: data_size|pad fields, tensor_dim0, tile_dim0/1, dim0 stride).

#if USE_TDM
DEVI void tdm_load_2d(unsigned lds_addr, const void* gptr,
                      unsigned tile_k, unsigned tile_rows,
                      unsigned stride_elems,
                      unsigned pad_int_code, unsigned pad_amt_code) {
  const unsigned long long ga = (unsigned long long)gptr;
  v4u g0;
  g0[0] = 1u;                                  // count=1 (valid descriptor)
  g0[1] = lds_addr;                            // lds_addr (bytes)
  g0[2] = (unsigned)ga;                        // global_addr[31:0]
  g0[3] = (unsigned)(ga >> 32) | (2u << 30);   // global_addr[56:32] | type=2

  const unsigned td0 = stride_elems;           // tensor_dim0 (elements)
  const unsigned td1 = 0x7fffffffu;            // tensor_dim1: effectively inf
  v8i g1;
  g1[0] = (int)((1u << 16)                     // data_size = 2 bytes
              | (1u << 20)                     // pad_enable
              | (pad_int_code << 22)           // pad_interval
              | (pad_amt_code << 25));         // pad_amount
  g1[1] = (int)((td0 & 0xffffu) << 16);        // tensor_dim0[15:0]
  g1[2] = (int)((td0 >> 16) | ((td1 & 0xffffu) << 16));
  g1[3] = (int)((td1 >> 16) | (tile_k << 16)); // tile_dim0 = tile_k
  g1[4] = (int)(tile_rows & 0xffffu);          // tile_dim1, tile_dim2 = 0
  g1[5] = (int)stride_elems;                   // tensor_dim0_stride[31:0]
  g1[6] = 0;                                   // stride hi | dim1_stride lo
  g1[7] = 0;
  v4i g2 = {0, 0, 0, 0};
  v4i g3 = {0, 0, 0, 0};
#if __clang_major__ >= 23
  v8i g4 = {0, 0, 0, 0, 0, 0, 0, 0};
  __builtin_amdgcn_tensor_load_to_lds(g0, g1, g2, g3, g4, 0);
#else
  __builtin_amdgcn_tensor_load_to_lds(g0, g1, g2, g3, 0);
#endif
}
#endif  // USE_TDM

// ---------------------------- elementwise kernels ---------------------------

__global__ void f32_to_f16_kernel(const float* __restrict__ in,
                                  _Float16* __restrict__ out, long n) {
  for (long i = (long)blockIdx.x * blockDim.x + threadIdx.x; i < n;
       i += (long)gridDim.x * blockDim.x)
    out[i] = (_Float16)in[i];
}

__global__ void copy_f32_kernel(const float* __restrict__ in,
                                float* __restrict__ out, long n) {
  for (long i = (long)blockIdx.x * blockDim.x + threadIdx.x; i < n;
       i += (long)gridDim.x * blockDim.x)
    out[i] = in[i];
}

// ------------------------------- LayerNorm ----------------------------------

template <typename Tin>
__global__ __launch_bounds__(256)
void ln_kernel(const Tin* __restrict__ in, const float* __restrict__ g,
               const float* __restrict__ b, _Float16* __restrict__ out,
               int cols) {
  __shared__ float smem[16];
  const long row = blockIdx.x;
  const Tin* p = in + row * (long)cols;
  float s = 0.f, ss = 0.f;
  for (int c = threadIdx.x; c < cols; c += 256) {
    float v = (float)p[c];
    s += v; ss += v * v;
  }
#pragma unroll
  for (int m = 16; m >= 1; m >>= 1) {
    s  += __shfl_xor(s,  m, 32);
    ss += __shfl_xor(ss, m, 32);
  }
  const int wid = threadIdx.x >> 5, lane = threadIdx.x & 31;
  if (lane == 0) { smem[wid] = s; smem[8 + wid] = ss; }
  __syncthreads();
  if (threadIdx.x == 0) {
    float ts = 0.f, tss = 0.f;
#pragma unroll
    for (int i = 0; i < 8; ++i) { ts += smem[i]; tss += smem[8 + i]; }
    smem[0] = ts; smem[8] = tss;
  }
  __syncthreads();
  const float mean = smem[0] / cols;
  const float var  = smem[8] / cols - mean * mean;
  const float rstd = rsqrtf(var + 1e-5f);
  for (int c = threadIdx.x; c < cols; c += 256) {
    float v = (float)p[c];
    out[row * (long)cols + c] = (_Float16)((v - mean) * rstd * g[c] + b[c]);
  }
}

// --------------------------------- GEMM -------------------------------------
// Out[M,Nout] = epilogue( A[M,K] @ W[Nout,K]^T + bias ), f16 inputs, f32 acc.
// 256 threads = 8 waves; 128x128 tile; K-step 32 == one WMMA depth.
// Wave grid 2(M) x 4(N): each wave owns 64x32 = 4x2 WMMA accumulators.
// LDS rows padded 32->40 f16 (TDM descriptor pad: 16 dwords + 4 dwords).
// Double-buffered: TDM pair for tile t+1 is issued before waiting on tile t
// (TENSORcnt retires in order), overlapping DMA with the WMMA chain.

template <bool OUT16, bool BIAS, bool GELU, bool ACCUM>
__global__ __launch_bounds__(256)
void gemm_kernel(const _Float16* __restrict__ A, const _Float16* __restrict__ W,
                 const float* __restrict__ bias, void* __restrict__ OutV,
                 int M, int Nout, int K) {
  __shared__ __align__(16) _Float16 As[2][128][40];
  __shared__ __align__(16) _Float16 Ws[2][128][40];

  const int tid  = threadIdx.x;
  const int lane = tid & 31;
  const int wid  = tid >> 5;
  const int m0   = blockIdx.y * 128;
  const int n0   = blockIdx.x * 128;
  const int wm   = (wid >> 2) * 64;   // 0 / 64
  const int wn   = (wid & 3) * 32;    // 0,32,64,96

  v8f acc[4][2];
  const v8f vzero = {0.f, 0.f, 0.f, 0.f, 0.f, 0.f, 0.f, 0.f};
#pragma unroll
  for (int i = 0; i < 4; ++i)
#pragma unroll
    for (int j = 0; j < 2; ++j) acc[i][j] = vzero;

  const _Float16* gA0 = A + (long)m0 * K;
  const _Float16* gW0 = W + (long)n0 * K;
  const int ntiles = K / 32;

#if USE_TDM
  if (wid == 0) {   // prologue: DMA tile 0 into buffer 0
    tdm_load_2d((unsigned)(size_t)&As[0][0][0], gA0, 32, 128, (unsigned)K, 3, 3);
    tdm_load_2d((unsigned)(size_t)&Ws[0][0][0], gW0, 32, 128, (unsigned)K, 3, 3);
  }
#else
  const int lrow = tid >> 1;          // 0..127
  const int lseg = (tid & 1) * 16;    // 0 / 16 (f16 elements)
  const _Float16* gArow = gA0 + (long)lrow * K + lseg;
  const _Float16* gWrow = gW0 + (long)lrow * K + lseg;
#endif

  for (int t = 0; t < ntiles; ++t) {
    const int cur = t & 1;
    __syncthreads();   // compute of tile t-1 done: buffers safe to touch
#if USE_TDM
    if (wid == 0) {
      if (t + 1 < ntiles) {   // prefetch next tile into the other buffer
        const int nxt = (t + 1) & 1;
        tdm_load_2d((unsigned)(size_t)&As[nxt][0][0], gA0 + (t + 1) * 32,
                    32, 128, (unsigned)K, 3, 3);
        tdm_load_2d((unsigned)(size_t)&Ws[nxt][0][0], gW0 + (t + 1) * 32,
                    32, 128, (unsigned)K, 3, 3);
        __builtin_amdgcn_s_wait_tensorcnt(2);  // tile t retired (in-order)
      } else {
        __builtin_amdgcn_s_wait_tensorcnt(0);
      }
    }
#else
    {
      const int k0 = t * 32;
      const uint4* ga = (const uint4*)(gArow + k0);
      *(uint4*)&As[cur][lrow][lseg]     = ga[0];
      *(uint4*)&As[cur][lrow][lseg + 8] = ga[1];
      const uint4* gw = (const uint4*)(gWrow + k0);
      *(uint4*)&Ws[cur][lrow][lseg]     = gw[0];
      *(uint4*)&Ws[cur][lrow][lseg + 8] = gw[1];
      if (k0 + 32 < K) {                      // gfx1250 global_prefetch_b8
        __builtin_prefetch(gArow + k0 + 32, 0, 0);
        __builtin_prefetch(gWrow + k0 + 32, 0, 0);
      }
    }
#endif
    __syncthreads();

    // Preload every fragment, then stream 8 back-to-back WMMAs.
    v16h bfr0 = ld_bfrag(&Ws[cur][wn][0],      40, lane);
    v16h bfr1 = ld_bfrag(&Ws[cur][wn + 16][0], 40, lane);
    v16h af[4];
#pragma unroll
    for (int i = 0; i < 4; ++i)
      af[i] = ld_afrag(&As[cur][wm + i * 16][0], 40, lane);
#pragma unroll
    for (int i = 0; i < 4; ++i) {
      acc[i][0] = wmma_f16(af[i], bfr0, acc[i][0]);
      acc[i][1] = wmma_f16(af[i], bfr1, acc[i][1]);
    }
  }

  // Epilogue. C/D layout: VGPR r -> row (lane>>4)*8 + r, col = lane&15.
  const int crow0 = m0 + wm + ((lane >> 4) << 3);
  const int ccol0 = n0 + wn + (lane & 15);
#pragma unroll
  for (int i = 0; i < 4; ++i)
#pragma unroll
    for (int j = 0; j < 2; ++j) {
      const int col = ccol0 + j * 16;
#pragma unroll
      for (int r = 0; r < 8; ++r) {
        const long row = crow0 + i * 16 + r;
        float val = acc[i][j][r];
        if (BIAS) val += bias[col];
        if (GELU) val = gelu_exact(val);
        const long idx = row * (long)Nout + col;
        if (OUT16) {
          ((_Float16*)OutV)[idx] = (_Float16)val;
        } else {
          float* O = (float*)OutV;
          O[idx] = ACCUM ? (O[idx] + val) : val;
        }
      }
    }
}

// ---------------------------- Flash attention -------------------------------
// One block = 64 query rows (4 waves x 16 rows), HD = 64, key tiles of 64.
// S = Q K^T via WMMA, online softmax (shfl width-16 row reductions), P staged
// through LDS and re-read as A-fragments, V staged transposed for B-fragments.
// Q and double-buffered K tiles ride the TDM (pad 32+4 dwords -> 72 stride);
// K(jb+1) DMA overlaps the compute of K(jb).

__global__ __launch_bounds__(128)
void attn_kernel(const _Float16* __restrict__ Q, int qs, int qc,
                 const _Float16* __restrict__ Kp, int ks, int kc,
                 const _Float16* __restrict__ Vp, int vs, int vc,
                 _Float16* __restrict__ O, int os, int oc,
                 int nkeys, int causal) {
  __shared__ __align__(16) _Float16 Qs [64][72];
  __shared__ __align__(16) _Float16 Ksb[2][64][72];
  __shared__ __align__(16) _Float16 Vts[64][72];   // transposed: [d][key]
  __shared__ __align__(16) _Float16 Ps [64][72];

  const int tid  = threadIdx.x;
  const int lane = tid & 31;
  const int w    = tid >> 5;                 // wave 0..3 -> query rows w*16..
  const int bh   = blockIdx.y;
  const int b    = bh / H_;
  const int h    = bh % H_;
  const int q0   = blockIdx.x * 64;
  const long rowbase = (long)b * N_;
  const int hcol = h * HD_;
  const int jend = causal ? blockIdx.x : (nkeys / 64 - 1);

#if USE_TDM
  if (w == 0) {   // Q tile + first K tile; completion covered by first wait
    tdm_load_2d((unsigned)(size_t)&Qs[0][0],
                Q + (rowbase + q0) * (long)qs + qc + hcol, 64, 64,
                (unsigned)qs, 4, 3);
    tdm_load_2d((unsigned)(size_t)&Ksb[0][0][0],
                Kp + rowbase * (long)ks + kc + hcol, 64, 64,
                (unsigned)ks, 4, 3);
  }
#else
  {
    const int r = tid >> 1, seg = (tid & 1) * 32;
    const uint4* gq =
        (const uint4*)(Q + (rowbase + q0 + r) * (long)qs + qc + hcol + seg);
    uint4* dq = (uint4*)&Qs[r][seg];
    dq[0] = gq[0]; dq[1] = gq[1]; dq[2] = gq[2]; dq[3] = gq[3];
  }
#endif

  float m_i[8], l_i[8];
  v8f o_acc[4];
  const v8f vzero = {0.f, 0.f, 0.f, 0.f, 0.f, 0.f, 0.f, 0.f};
#pragma unroll
  for (int r = 0; r < 8; ++r) { m_i[r] = -1e30f; l_i[r] = 0.f; }
#pragma unroll
  for (int j = 0; j < 4; ++j) o_acc[j] = vzero;

  const int half8 = (lane >> 4) * 8;
  const int nbase = lane & 15;

  for (int jb = 0; jb <= jend; ++jb) {
    const int cur = jb & 1;
    __syncthreads();   // previous iteration's compute fully retired
#if USE_TDM
    if (w == 0 && jb + 1 <= jend)   // prefetch next K tile into other buffer
      tdm_load_2d((unsigned)(size_t)&Ksb[(jb + 1) & 1][0][0],
                  Kp + (rowbase + (jb + 1) * 64) * (long)ks + kc + hcol,
                  64, 64, (unsigned)ks, 4, 3);
#else
    {
      const int r = tid >> 1, seg = (tid & 1) * 32;
      const uint4* gk = (const uint4*)(Kp + (rowbase + jb * 64 + r) * (long)ks +
                                       kc + hcol + seg);
      uint4* dk = (uint4*)&Ksb[cur][r][seg];
      dk[0] = gk[0]; dk[1] = gk[1]; dk[2] = gk[2]; dk[3] = gk[3];
    }
#endif
    // V tile transposed into Vts[d][key] so P@V B-fragments are contiguous.
#pragma unroll
    for (int it = 0; it < 8; ++it) {
      const int q2  = tid + it * 128;        // 0..1023
      const int key = q2 >> 4;
      const int d4  = (q2 & 15) * 4;
      const _Float16* gv =
          Vp + (rowbase + jb * 64 + key) * (long)vs + vc + hcol + d4;
      float2 t2 = *(const float2*)gv;        // 4 f16
      const _Float16* t = (const _Float16*)&t2;
      Vts[d4 + 0][key] = t[0]; Vts[d4 + 1][key] = t[1];
      Vts[d4 + 2][key] = t[2]; Vts[d4 + 3][key] = t[3];
    }
#if USE_TDM
    if (w == 0) {
      if (jb < jend) __builtin_amdgcn_s_wait_tensorcnt(1);  // K(jb) retired
      else           __builtin_amdgcn_s_wait_tensorcnt(0);
    }
#endif
    __syncthreads();

    // ---- S = Q K^T (64x64 per block, 16x64 per wave) ----
    v8f s[4];
#pragma unroll
    for (int j = 0; j < 4; ++j) s[j] = vzero;
#pragma unroll
    for (int kk = 0; kk < 64; kk += 32) {
      v16h aq = ld_afrag(&Qs[w * 16][kk], 72, lane);
      v16h bk[4];
#pragma unroll
      for (int j = 0; j < 4; ++j)
        bk[j] = ld_bfrag(&Ksb[cur][j * 16][kk], 72, lane);
#pragma unroll
      for (int j = 0; j < 4; ++j) s[j] = wmma_f16(aq, bk[j], s[j]);
    }

    // ---- online softmax, write P tile ----
#pragma unroll
    for (int r = 0; r < 8; ++r) {
      const int row_l = w * 16 + half8 + r;
      const int qi    = q0 + row_l;
      float sv[4];
      float mnew = m_i[r];
#pragma unroll
      for (int j = 0; j < 4; ++j) {
        const int col = jb * 64 + j * 16 + nbase;
        float x = s[j][r] * SM_SCALE;
        if (causal && col > qi) x = -1e30f;
        sv[j] = x;
        mnew  = fmaxf(mnew, x);
      }
#pragma unroll
      for (int mm = 8; mm >= 1; mm >>= 1)
        mnew = fmaxf(mnew, __shfl_xor(mnew, mm, 16));
      const float alpha = __expf(m_i[r] - mnew);
      float ls = 0.f;
#pragma unroll
      for (int j = 0; j < 4; ++j) {
        const float pj = __expf(sv[j] - mnew);
        ls += pj;
        Ps[row_l][j * 16 + nbase] = (_Float16)pj;
      }
#pragma unroll
      for (int mm = 8; mm >= 1; mm >>= 1) ls += __shfl_xor(ls, mm, 16);
      l_i[r] = l_i[r] * alpha + ls;
      m_i[r] = mnew;
#pragma unroll
      for (int jd = 0; jd < 4; ++jd) o_acc[jd][r] *= alpha;
    }
    __syncthreads();

    // ---- O += P V ----
#pragma unroll
    for (int kk = 0; kk < 64; kk += 32) {
      v16h ap = ld_afrag(&Ps[w * 16][kk], 72, lane);
      v16h bv[4];
#pragma unroll
      for (int jd = 0; jd < 4; ++jd)
        bv[jd] = ld_bfrag(&Vts[jd * 16][kk], 72, lane);
#pragma unroll
      for (int jd = 0; jd < 4; ++jd)
        o_acc[jd] = wmma_f16(ap, bv[jd], o_acc[jd]);
    }
  }

  // ---- normalize and store ----
#pragma unroll
  for (int r = 0; r < 8; ++r) {
    const int row_l = w * 16 + half8 + r;
    const float inv = 1.0f / l_i[r];
    const long obase = (rowbase + q0 + row_l) * (long)os + oc + hcol;
#pragma unroll
    for (int jd = 0; jd < 4; ++jd)
      O[obase + jd * 16 + nbase] = (_Float16)(o_acc[jd][r] * inv);
  }
}

// ------------------------------ launcher ------------------------------------

extern "C" void kernel_launch(void* const* d_in, const int* in_sizes, int n_in,
                              void* d_out, int out_size, void* d_ws,
                              size_t ws_size, hipStream_t stream) {
  (void)in_sizes; (void)n_in; (void)out_size; (void)ws_size;

  const float* x        = (const float*)d_in[0];
  const float* y        = (const float*)d_in[1];
  /* d_in[2] = mask : causal mask computed analytically, not read */
  const float* a1_ln1_g = (const float*)d_in[3];
  const float* a1_ln1_b = (const float*)d_in[4];
  const float* a1_qk_w  = (const float*)d_in[5];
  const float* a1_v_w   = (const float*)d_in[6];
  const float* a1_ln2_g = (const float*)d_in[7];
  const float* a1_ln2_b = (const float*)d_in[8];
  const float* a1_pj_w  = (const float*)d_in[9];
  const float* a1_pj_b  = (const float*)d_in[10];
  const float* a2_ln_g  = (const float*)d_in[11];
  const float* a2_ln_b  = (const float*)d_in[12];
  const float* a2_q_w   = (const float*)d_in[13];
  const float* a2_kv_w  = (const float*)d_in[14];
  const float* a2_pj_w  = (const float*)d_in[15];
  const float* a2_pj_b  = (const float*)d_in[16];
  const float* m_ln1_g  = (const float*)d_in[17];
  const float* m_ln1_b  = (const float*)d_in[18];
  const float* m_fc1_w  = (const float*)d_in[19];
  const float* m_fc1_b  = (const float*)d_in[20];
  const float* m_ln2_g  = (const float*)d_in[21];
  const float* m_ln2_b  = (const float*)d_in[22];
  const float* m_fc2_w  = (const float*)d_in[23];
  const float* m_fc2_b  = (const float*)d_in[24];

  float* xcur = (float*)d_out;   // f32 residual stream lives in d_out

  // f16 scratch carve-out
  _Float16* ws = (_Float16*)d_ws;
  size_t off = 0;
  auto carve = [&](size_t n) { _Float16* p = ws + off; off += n; return p; };
  _Float16* w_qk  = carve((size_t)2 * C_ * C_);
  _Float16* w_v   = carve((size_t)C_ * C_);
  _Float16* w_p1  = carve((size_t)C_ * C_);
  _Float16* w_q2  = carve((size_t)C_ * C_);
  _Float16* w_kv  = carve((size_t)2 * C_ * Y_);
  _Float16* w_p2  = carve((size_t)C_ * C_);
  _Float16* w_fc1 = carve((size_t)HID_ * C_);
  _Float16* w_fc2 = carve((size_t)C_ * HID_);
  _Float16* y16   = carve((size_t)M_ * Y_);
  _Float16* h16   = carve((size_t)M_ * C_);
  _Float16* qk16  = carve((size_t)M_ * 2 * C_);  // also reused as kv16
  _Float16* v16   = carve((size_t)M_ * C_);
  _Float16* q16   = carve((size_t)M_ * C_);
  _Float16* o16   = carve((size_t)M_ * C_);
  _Float16* hid16 = carve((size_t)M_ * HID_);
  _Float16* hid2  = carve((size_t)M_ * HID_);

  auto cvt = [&](const float* src, _Float16* dst, long n) {
    f32_to_f16_kernel<<<2048, 256, 0, stream>>>(src, dst, n);
  };

  // ---- one-time (per launch) conversions ----
  copy_f32_kernel<<<2048, 256, 0, stream>>>(x, xcur, (long)M_ * C_);
  cvt(a1_qk_w, w_qk,  (long)2 * C_ * C_);
  cvt(a1_v_w,  w_v,   (long)C_ * C_);
  cvt(a1_pj_w, w_p1,  (long)C_ * C_);
  cvt(a2_q_w,  w_q2,  (long)C_ * C_);
  cvt(a2_kv_w, w_kv,  (long)2 * C_ * Y_);
  cvt(a2_pj_w, w_p2,  (long)C_ * C_);
  cvt(m_fc1_w, w_fc1, (long)HID_ * C_);
  cvt(m_fc2_w, w_fc2, (long)C_ * HID_);
  cvt(y,       y16,   (long)M_ * Y_);

  const dim3 blk256(256), blk128(128);
  auto ggrid = [](int Nout) { return dim3(Nout / 128, M_ / 128); };
  const dim3 agrid(N_ / 64, B_ * H_);

  // ================= self-attention =================
  ln_kernel<float><<<M_, blk256, 0, stream>>>(xcur, a1_ln1_g, a1_ln1_b, h16, C_);
  gemm_kernel<true, false, false, false><<<ggrid(2 * C_), blk256, 0, stream>>>(
      h16, w_qk, nullptr, qk16, M_, 2 * C_, C_);
  gemm_kernel<true, false, false, false><<<ggrid(C_), blk256, 0, stream>>>(
      h16, w_v, nullptr, v16, M_, C_, C_);
  attn_kernel<<<agrid, blk128, 0, stream>>>(
      qk16, 2 * C_, 0,      /* q  */
      qk16, 2 * C_, C_,     /* k  */
      v16,  C_,     0,      /* v  */
      o16,  C_,     0,      /* o  */
      N_, /*causal=*/1);
  ln_kernel<_Float16><<<M_, blk256, 0, stream>>>(o16, a1_ln2_g, a1_ln2_b, h16, C_);
  gemm_kernel<false, true, false, true><<<ggrid(C_), blk256, 0, stream>>>(
      h16, w_p1, a1_pj_b, xcur, M_, C_, C_);   // xcur += proj(ln2(o))

  // ================= cross-attention =================
  ln_kernel<float><<<M_, blk256, 0, stream>>>(xcur, a2_ln_g, a2_ln_b, h16, C_);
  gemm_kernel<true, false, false, false><<<ggrid(C_), blk256, 0, stream>>>(
      h16, w_q2, nullptr, q16, M_, C_, C_);
  gemm_kernel<true, false, false, false><<<ggrid(2 * C_), blk256, 0, stream>>>(
      y16, w_kv, nullptr, qk16, M_, 2 * C_, Y_);     // kv lives in qk16
  attn_kernel<<<agrid, blk128, 0, stream>>>(
      q16,  C_,     0,      /* q  */
      qk16, 2 * C_, 0,      /* k  */
      qk16, 2 * C_, C_,     /* v  */
      o16,  C_,     0,      /* o  */
      N_, /*causal=*/0);
  gemm_kernel<false, true, false, true><<<ggrid(C_), blk256, 0, stream>>>(
      o16, w_p2, a2_pj_b, xcur, M_, C_, C_);   // xcur += proj(o)

  // ================= MLP =================
  ln_kernel<float><<<M_, blk256, 0, stream>>>(xcur, m_ln1_g, m_ln1_b, h16, C_);
  gemm_kernel<true, true, true, false><<<ggrid(HID_), blk256, 0, stream>>>(
      h16, w_fc1, m_fc1_b, hid16, M_, HID_, C_);     // gelu(fc1 + b)
  ln_kernel<_Float16><<<M_, blk256, 0, stream>>>(hid16, m_ln2_g, m_ln2_b, hid2,
                                                 HID_);
  gemm_kernel<false, true, false, true><<<ggrid(C_), blk256, 0, stream>>>(
      hid2, w_fc2, m_fc2_b, xcur, M_, C_, HID_);     // xcur += fc2 + b
}
